// RotaryMultiheadAttention_13889924235389
// MI455X (gfx1250) — matrix operand order repaired and checked by reading, not compiled, not run on hardware
//
#include <hip/hip_runtime.h>
#include <hip/hip_bf16.h>
#include <math.h>

#define E_  1024
#define H_  16
#define D_  64
#define B_  2
#define S_  2048
#define BS_ (B_ * S_)          // 4096 rows of the (B*S, E) activations

typedef __attribute__((ext_vector_type(16))) _Float16 v16h;
typedef __attribute__((ext_vector_type(8)))  float    v8f;
typedef __attribute__((ext_vector_type(4)))  unsigned int u32x4;
typedef __attribute__((ext_vector_type(8)))  int      i32x8;
typedef __attribute__((ext_vector_type(4)))  int      i32x4;

#if defined(__has_builtin)
#if __has_builtin(__builtin_amdgcn_tensor_load_to_lds) && __has_builtin(__builtin_amdgcn_s_wait_tensorcnt)
#define USE_TDM 1
#endif
#endif

// ---------------------------------------------------------------------------
// Fragment loaders: contiguous k order (k = 16h + s), same in A and B, not the ISA table order
// A-matrix 16x32 f16: lanes 0-15 row M=lane, K=0..15; lanes 16-31
// same rows, K=16..31. Two contiguous 16B chunks per lane.
// ---------------------------------------------------------------------------
__device__ __forceinline__ v16h ld_a_f16(const _Float16* __restrict__ base, int ld) {
    const int lane = threadIdx.x & 31;
    const _Float16* p = base + (size_t)(lane & 15) * ld + ((lane >> 4) << 4);
    v16h f;
    *((float4*)&f)       = *(const float4*)(p);        // K = kb .. kb+7
    *(((float4*)&f) + 1) = *(const float4*)(p + 8);    // K = kb+8 .. kb+15
    return f;
}

__device__ __forceinline__ v16h ld_a_f32(const float* __restrict__ base, int ld) {
    const int lane = threadIdx.x & 31;
    const float* p = base + (size_t)(lane & 15) * ld + ((lane >> 4) << 4);
    float t[16];
    *(float4*)(t + 0)  = *(const float4*)(p + 0);
    *(float4*)(t + 4)  = *(const float4*)(p + 4);
    *(float4*)(t + 8)  = *(const float4*)(p + 8);
    *(float4*)(t + 12) = *(const float4*)(p + 12);
    v16h f;
#pragma unroll
    for (int i = 0; i < 16; ++i) f[i] = (_Float16)t[i];
    return f;
}

// B-matrix 32x16 f16 (B[k][n] = W[n][k], W row-major K-contiguous):
// lane = column n (mod 16); lanes 0-15 hold K=0..15, lanes 16-31 K=16..31.
__device__ __forceinline__ v16h ld_b_f16(const _Float16* __restrict__ base, int ld) {
    const int lane = threadIdx.x & 31;
    const _Float16* p = base + (size_t)(lane & 15) * ld + ((lane >> 4) << 4);
    v16h f;
    *((float4*)&f)       = *(const float4*)(p);
    *(((float4*)&f) + 1) = *(const float4*)(p + 8);
    return f;
}

__device__ __forceinline__ v16h ld_b_f32(const float* __restrict__ base, int ld) {
    const int lane = threadIdx.x & 31;
    const float* p = base + (size_t)(lane & 15) * ld + ((lane >> 4) << 4);
    float t[16];
    *(float4*)(t + 0)  = *(const float4*)(p + 0);
    *(float4*)(t + 4)  = *(const float4*)(p + 4);
    *(float4*)(t + 8)  = *(const float4*)(p + 8);
    *(float4*)(t + 12) = *(const float4*)(p + 12);
    v16h f;
#pragma unroll
    for (int i = 0; i < 16; ++i) f[i] = (_Float16)t[i];
    return f;
}

// C/D 16x16 f32: VGPR r -> M = r + 8*(lane>=16), N = lane & 15
__device__ __forceinline__ void st_c_f32(float* __restrict__ base, int ld, v8f c,
                                         float scale, const float* __restrict__ bias) {
    const int lane = threadIdx.x & 31;
    const int n  = lane & 15;
    const int mo = (lane >> 4) << 3;
#pragma unroll
    for (int r = 0; r < 8; ++r) {
        float v = c[r] * scale;
        if (bias) v += bias[n];
        base[(size_t)(mo + r) * ld + n] = v;
    }
}

__device__ __forceinline__ void st_c_f16(_Float16* __restrict__ base, int ld, v8f c) {
    const int lane = threadIdx.x & 31;
    const int n  = lane & 15;
    const int mo = (lane >> 4) << 3;
#pragma unroll
    for (int r = 0; r < 8; ++r)
        base[(size_t)(mo + r) * ld + n] = (_Float16)c[r];
}

__device__ __forceinline__ void wmma_acc(v8f (&acc)[2][2], const v16h a[2], const v16h b[2]) {
#pragma unroll
    for (int i = 0; i < 2; ++i)
#pragma unroll
        for (int j = 0; j < 2; ++j)
            acc[i][j] = __builtin_amdgcn_wmma_f32_16x16x32_f16(
                false, a[i], false, b[j], (short)0, acc[i][j], false, false);
}

#ifdef USE_TDM
// ---------------------------------------------------------------------------
// Tensor Data Mover: one descriptor moves a rows x 64 f16 2D tile (row stride
// D_=64 elements) from global into LDS. D# layout per CDNA5 ISA.
// ---------------------------------------------------------------------------
__device__ __forceinline__ void tdm_load_tile_f16(unsigned lds_off,
                                                  unsigned long long gaddr,
                                                  int rows) {
    u32x4 g0;
    g0[0] = 1u;                                             // count=1, user mode
    g0[1] = lds_off;                                        // lds_addr (bytes)
    g0[2] = (unsigned)(gaddr & 0xFFFFFFFFu);                // global_addr[31:0]
    g0[3] = (unsigned)((gaddr >> 32) & 0x01FFFFFFu) | 0x80000000u; // addr[56:32], type=2
    i32x8 g1;
    g1[0] = (1 << 16);          // data_size = 1 -> 2 bytes/elem
    g1[1] = (D_ << 16);         // tensor_dim0 = 64  (bits 79:48, low half)
    g1[2] = (S_ << 16);         // tensor_dim1 = 2048 (bits 111:80, low half)
    g1[3] = (D_ << 16);         // tile_dim0 = 64    (bits 127:112)
    g1[4] = rows;               // tile_dim1 = rows  (bits 143:128); tile_dim2=0
    g1[5] = D_;                 // tensor_dim0_stride[31:0] = 64
    g1[6] = 0;                  // stride msbs / dim1_stride lo
    g1[7] = 0;
    i32x4 g2 = {0, 0, 0, 0};
    i32x4 g3 = {0, 0, 0, 0};
#if defined(__clang_major__) && (__clang_major__ >= 23)
    i32x8 g4 = {0, 0, 0, 0, 0, 0, 0, 0};
    __builtin_amdgcn_tensor_load_to_lds(g0, g1, g2, g3, g4, 0);
#else
    __builtin_amdgcn_tensor_load_to_lds(g0, g1, g2, g3, 0);
#endif
}
#endif

// ---------------------------------------------------------------------------
// Kernel 1: projection GEMM  Y[M,N] = X[M,K] @ W[N,K]^T   (fp32 in, fp32 out)
// M=BS_=4096, N=E_=1024, K=E_=1024. 256 thr = 8 waves; block tile 128x64.
// ---------------------------------------------------------------------------
__global__ void __launch_bounds__(256)
gemm_xw_kernel(const float* __restrict__ X, const float* __restrict__ W,
               float* __restrict__ Y) {
    const int wid = threadIdx.x >> 5;
    const int m0 = blockIdx.x * 128 + (wid >> 1) * 32;
    const int n0 = blockIdx.y * 64  + (wid & 1) * 32;
    v8f acc[2][2] = {};
    for (int k0 = 0; k0 < E_; k0 += 32) {
        __builtin_prefetch(X + (size_t)m0 * E_ + k0 + 64, 0, 0);   // global_prefetch_b8
        v16h a[2], b[2];
        a[0] = ld_a_f32(X + (size_t)(m0 +  0) * E_ + k0, E_);
        a[1] = ld_a_f32(X + (size_t)(m0 + 16) * E_ + k0, E_);
        b[0] = ld_b_f32(W + (size_t)(n0 +  0) * E_ + k0, E_);
        b[1] = ld_b_f32(W + (size_t)(n0 + 16) * E_ + k0, E_);
        wmma_acc(acc, a, b);
    }
#pragma unroll
    for (int i = 0; i < 2; ++i)
#pragma unroll
        for (int j = 0; j < 2; ++j)
            st_c_f32(Y + (size_t)(m0 + i * 16) * E_ + n0 + j * 16, E_,
                     acc[i][j], 1.0f, nullptr);
}

// ---------------------------------------------------------------------------
// Kernel 2: bias + RoPE, fp32 [BS,E] -> f16 [B,H,S,D]
// ---------------------------------------------------------------------------
__global__ void __launch_bounds__(256)
rope_kernel(const float* __restrict__ Y, const float* __restrict__ bias,
            _Float16* __restrict__ out) {
    const int idx = blockIdx.x * 256 + threadIdx.x;     // over B*H*S*D = 4194304
    const int d  = idx & (D_ - 1);
    const int s  = (idx >> 6) & (S_ - 1);
    const int bh = idx >> 17;
    const int h = bh & (H_ - 1);
    const int b = bh >> 4;
    const int row = b * S_ + s;
    const int j  = d & 31;
    const int d2 = (d < 32) ? d + 32 : d - 32;
    const float y  = Y[(size_t)row * E_ + h * D_ + d]  + bias[h * D_ + d];
    const float y2 = Y[(size_t)row * E_ + h * D_ + d2] + bias[h * D_ + d2];
    // inv_freq[j] = 10000^(-j/32);  ln(10000) = 9.210340371976184
    const float ang = (float)s * __expf(-(float)j * (9.210340371976184f / 32.0f));
    const float c = __cosf(ang), sn = __sinf(ang);
    const float r = y * c + ((d < 32) ? -y2 : y2) * sn;
    out[idx] = (_Float16)r;
}

// ---------------------------------------------------------------------------
// Kernel 3: bias + transpose, fp32 [BS,E] -> f16 [B,H,D,S]  (V for PV GEMM)
// ---------------------------------------------------------------------------
__global__ void __launch_bounds__(256)
vtrans_kernel(const float* __restrict__ V, const float* __restrict__ bias,
              _Float16* __restrict__ out) {
    const int idx = blockIdx.x * 256 + threadIdx.x;     // over B*H*D*S
    const int s  = idx & (S_ - 1);
    const int d  = (idx >> 11) & (D_ - 1);
    const int bh = idx >> 17;
    const int h = bh & (H_ - 1);
    const int b = bh >> 4;
    const float v = V[(size_t)(b * S_ + s) * E_ + h * D_ + d] + bias[h * D_ + d];
    out[idx] = (_Float16)v;
}

// ---------------------------------------------------------------------------
// Kernel 4: scores = (Q_h @ K_h^T) * 1/sqrt(D), f16 in, fp32 out (B,H,S,S)
// Q (128x64) and K (64x64) tiles staged in LDS via the Tensor Data Mover;
// wave 0 issues both descriptors, waits on TENSORcnt, barrier publishes LDS.
// ---------------------------------------------------------------------------
__global__ void __launch_bounds__(256)
attn_scores_kernel(const _Float16* __restrict__ Q, const _Float16* __restrict__ Kf,
                   float* __restrict__ attn) {
    extern __shared__ _Float16 smem[];          // 128*64 Q + 64*64 K = 24 KB
    _Float16* sQ = smem;                        // LDS byte offset 0
    _Float16* sK = smem + 128 * D_;             // LDS byte offset 16384
    const int bh = blockIdx.z;
    const int m0 = blockIdx.x * 128;
    const int n0 = blockIdx.y * 64;
    const _Float16* Qg = Q  + (size_t)bh * S_ * D_ + (size_t)m0 * D_;
    const _Float16* Kg = Kf + (size_t)bh * S_ * D_ + (size_t)n0 * D_;

#ifdef USE_TDM
    if ((threadIdx.x >> 5) == 0) {              // one wave drives the TDM
        tdm_load_tile_f16(0u,     (unsigned long long)(uintptr_t)Qg, 128);
        tdm_load_tile_f16(16384u, (unsigned long long)(uintptr_t)Kg, 64);
        __builtin_amdgcn_s_wait_tensorcnt(0);   // s_wait_tensorcnt 0
    }
#else
    // Fallback: cooperative global->LDS copy (128+64 rows of 64 f16 = 12288 h)
    for (int i = threadIdx.x; i < (128 + 64) * (D_ / 8); i += 256) {
        const int row = i >> 3, c8 = (i & 7) * 8;
        if (row < 128)
            *(float4*)(sQ + row * D_ + c8) = *(const float4*)(Qg + (size_t)row * D_ + c8);
        else
            *(float4*)(sK + (row - 128) * D_ + c8) =
                *(const float4*)(Kg + (size_t)(row - 128) * D_ + c8);
    }
#endif
    __syncthreads();

    const int wid = threadIdx.x >> 5;
    const int wm = (wid >> 1) * 32;
    const int wn = (wid & 1) * 32;
    v8f acc[2][2] = {};
#pragma unroll
    for (int k0 = 0; k0 < D_; k0 += 32) {
        v16h a[2], b[2];
        a[0] = ld_a_f16(sQ + (wm +  0) * D_ + k0, D_);
        a[1] = ld_a_f16(sQ + (wm + 16) * D_ + k0, D_);
        b[0] = ld_b_f16(sK + (wn +  0) * D_ + k0, D_);
        b[1] = ld_b_f16(sK + (wn + 16) * D_ + k0, D_);
        wmma_acc(acc, a, b);
    }
    const float scale = 0.125f;                 // 1/sqrt(64)
    float* Ch = attn + (size_t)bh * S_ * S_;
#pragma unroll
    for (int i = 0; i < 2; ++i)
#pragma unroll
        for (int j = 0; j < 2; ++j)
            st_c_f32(Ch + (size_t)(m0 + wm + i * 16) * S_ + n0 + wn + j * 16, S_,
                     acc[i][j], scale, nullptr);
}

// ---------------------------------------------------------------------------
// Kernel 5: in-place row softmax over the (B*H*S, S) score matrix
// ---------------------------------------------------------------------------
__global__ void __launch_bounds__(256)
softmax_kernel(float* __restrict__ attn) {
    __shared__ float red[256];
    float* p = attn + (size_t)blockIdx.x * S_;
    const int t = threadIdx.x;
    float x[8];
    float m = -3.4e38f;
#pragma unroll
    for (int i = 0; i < 8; ++i) { x[i] = p[i * 256 + t]; m = fmaxf(m, x[i]); }
    red[t] = m; __syncthreads();
    for (int off = 128; off > 0; off >>= 1) {
        if (t < off) red[t] = fmaxf(red[t], red[t + off]);
        __syncthreads();
    }
    m = red[0]; __syncthreads();
    float s = 0.f;
#pragma unroll
    for (int i = 0; i < 8; ++i) { x[i] = __expf(x[i] - m); s += x[i]; }
    red[t] = s; __syncthreads();
    for (int off = 128; off > 0; off >>= 1) {
        if (t < off) red[t] += red[t + off];
        __syncthreads();
    }
    const float inv = 1.0f / red[0];
#pragma unroll
    for (int i = 0; i < 8; ++i) p[i * 256 + t] = x[i] * inv;
}

// ---------------------------------------------------------------------------
// Kernel 6: attn_out = P @ V_h ; P fp32 (converted on the fly), Vt f16 [D,S]
// writes f16 into [B,S,E] layout (column = h*D + d) for the output GEMM
// ---------------------------------------------------------------------------
__global__ void __launch_bounds__(256)
attn_v_kernel(const float* __restrict__ attn, const _Float16* __restrict__ Vt,
              _Float16* __restrict__ AO) {
    const int bh = blockIdx.z;
    const int b = bh >> 4, h = bh & (H_ - 1);
    const int wid = threadIdx.x >> 5;
    const int m0 = blockIdx.x * 128 + (wid >> 1) * 32;
    const int n0 = (wid & 1) * 32;              // N = D_ = 64 total
    const float* P = attn + (size_t)bh * S_ * S_;
    const _Float16* Vh = Vt + (size_t)bh * D_ * S_;
    v8f acc[2][2] = {};
    for (int k0 = 0; k0 < S_; k0 += 32) {
        v16h a[2], bb[2];
        a[0]  = ld_a_f32(P + (size_t)(m0 +  0) * S_ + k0, S_);
        a[1]  = ld_a_f32(P + (size_t)(m0 + 16) * S_ + k0, S_);
        bb[0] = ld_b_f16(Vh + (size_t)(n0 +  0) * S_ + k0, S_);
        bb[1] = ld_b_f16(Vh + (size_t)(n0 + 16) * S_ + k0, S_);
        wmma_acc(acc, a, bb);
    }
    _Float16* C = AO + ((size_t)b * S_ + m0) * E_ + h * D_ + n0;
#pragma unroll
    for (int i = 0; i < 2; ++i)
#pragma unroll
        for (int j = 0; j < 2; ++j)
            st_c_f16(C + (size_t)(i * 16) * E_ + j * 16, E_, acc[i][j]);
}

// ---------------------------------------------------------------------------
// Kernel 7: output = AO[BS,E](f16) @ Wo[E,E]^T + bo  -> fp32 d_out
// ---------------------------------------------------------------------------
__global__ void __launch_bounds__(256)
gemm_ow_kernel(const _Float16* __restrict__ AO, const float* __restrict__ Wo,
               const float* __restrict__ bo, float* __restrict__ out) {
    const int wid = threadIdx.x >> 5;
    const int m0 = blockIdx.x * 128 + (wid >> 1) * 32;
    const int n0 = blockIdx.y * 64  + (wid & 1) * 32;
    v8f acc[2][2] = {};
    for (int k0 = 0; k0 < E_; k0 += 32) {
        v16h a[2], b[2];
        a[0] = ld_a_f16(AO + (size_t)(m0 +  0) * E_ + k0, E_);
        a[1] = ld_a_f16(AO + (size_t)(m0 + 16) * E_ + k0, E_);
        b[0] = ld_b_f32(Wo + (size_t)(n0 +  0) * E_ + k0, E_);
        b[1] = ld_b_f32(Wo + (size_t)(n0 + 16) * E_ + k0, E_);
        wmma_acc(acc, a, b);
    }
#pragma unroll
    for (int i = 0; i < 2; ++i)
#pragma unroll
        for (int j = 0; j < 2; ++j)
            st_c_f32(out + (size_t)(m0 + i * 16) * E_ + n0 + j * 16, E_,
                     acc[i][j], 1.0f, bo + n0 + j * 16);
}

// ---------------------------------------------------------------------------
extern "C" void kernel_launch(void* const* d_in, const int* in_sizes, int n_in,
                              void* d_out, int out_size, void* d_ws, size_t ws_size,
                              hipStream_t stream) {
    (void)in_sizes; (void)n_in; (void)out_size; (void)ws_size;
    const float* query = (const float*)d_in[0];
    const float* key   = (const float*)d_in[1];
    const float* value = (const float*)d_in[2];
    const float* Wq = (const float*)d_in[3];
    const float* bq = (const float*)d_in[4];
    const float* Wk = (const float*)d_in[5];
    const float* bk = (const float*)d_in[6];
    const float* Wv = (const float*)d_in[7];
    const float* bv = (const float*)d_in[8];
    const float* Wo = (const float*)d_in[9];
    const float* bo = (const float*)d_in[10];

    float* out  = (float*)d_out;                       // (B,S,E)
    float* attn = out + (size_t)BS_ * E_;              // (B,H,S,S)

    // Workspace carve-up (80 MB total):
    float* Qraw = (float*)d_ws;                        // 16 MB
    float* Kraw = Qraw + (size_t)BS_ * E_;             // 16 MB
    float* Vraw = Kraw + (size_t)BS_ * E_;             // 16 MB
    _Float16* Qh = (_Float16*)(Vraw + (size_t)BS_ * E_);  // 8 MB  [B,H,S,D]
    _Float16* Kh = Qh + (size_t)BS_ * E_;              // 8 MB  [B,H,S,D]
    _Float16* Vt = Kh + (size_t)BS_ * E_;              // 8 MB  [B,H,D,S]
    _Float16* AO = Vt + (size_t)BS_ * E_;              // 8 MB  [B,S,E]

    const dim3 blk(256);
    const dim3 gproj(BS_ / 128, E_ / 64);              // (32,16)
    gemm_xw_kernel<<<gproj, blk, 0, stream>>>(query, Wq, Qraw);
    gemm_xw_kernel<<<gproj, blk, 0, stream>>>(key,   Wk, Kraw);
    gemm_xw_kernel<<<gproj, blk, 0, stream>>>(value, Wv, Vraw);

    const dim3 gelem((BS_ * E_) / 256);                // 16384
    rope_kernel  <<<gelem, blk, 0, stream>>>(Qraw, bq, Qh);
    rope_kernel  <<<gelem, blk, 0, stream>>>(Kraw, bk, Kh);
    vtrans_kernel<<<gelem, blk, 0, stream>>>(Vraw, bv, Vt);

    attn_scores_kernel<<<dim3(S_ / 128, S_ / 64, B_ * H_), blk, 24576, stream>>>(Qh, Kh, attn);
    softmax_kernel    <<<dim3(B_ * H_ * S_), blk, 0, stream>>>(attn);
    attn_v_kernel     <<<dim3(S_ / 128, 1, B_ * H_), blk, 0, stream>>>(attn, Vt, AO);
    gemm_ow_kernel    <<<gproj, blk, 0, stream>>>(AO, Wo, bo, out);
}
